// MultiHeadFrequencyLayer_37469294690523
// MI455X (gfx1250) — matrix operand order repaired and checked by reading, not compiled, not run on hardware
//
#include <hip/hip_runtime.h>
#include <stdint.h>

// ---------------- problem constants ----------------
constexpr int B  = 4;
constexpr int S  = 8192;
constexpr int D  = 768;
constexpr int H  = 12;
constexpr int HD = 64;
constexpr int F  = S / 2 + 1;           // 4097
constexpr int KTOP = 409;               // int(F * 0.1)
constexpr float EPS = 1e-5f;
constexpr float ORTHO = 0.011048543456039806f;   // 1/sqrt(8192)

constexpr size_t NBDS = (size_t)B * D * S;       // 25,165,824
constexpr size_t NBDF = (size_t)B * D * F;       // 12,585,984
constexpr size_t BHF  = (size_t)B * H * F;       // 196,656

typedef __attribute__((ext_vector_type(16))) __bf16 v16bf;
typedef __attribute__((ext_vector_type(8)))  float  v8f;
typedef __attribute__((ext_vector_type(4)))  unsigned int u32x4;

struct U32x8 { u32x4 lo; u32x4 hi; };            // 32B carrier for v16bf bit_cast

// force global (AS1) vector loads so GEMM operands use global_load_b128,
// not flat_load_b128 (flat ticks DScnt and shares the LDS issue path)
typedef const __attribute__((address_space(1))) u32x4* gu4p;
__device__ __forceinline__ u32x4 gload16(const void* p) {
  return *(gu4p)(uintptr_t)p;
}

__device__ __forceinline__ __bf16 f2bf(float f) {
  unsigned u = __float_as_uint(f);
  u += 0x7fffu + ((u >> 16) & 1u);               // round-to-nearest-even
  unsigned short h = (unsigned short)(u >> 16);
  return __builtin_bit_cast(__bf16, h);
}

__device__ __forceinline__ float sigm(float x) {
  return 1.0f / (1.0f + __expf(-x));
}

// ================= 1) LayerNorm + transpose (B,S,D)->(B,D,S) =================
__global__ __launch_bounds__(256) void ln_kernel(const float* __restrict__ x,
                                                 const float* __restrict__ gamma,
                                                 const float* __restrict__ beta,
                                                 float* __restrict__ xnT) {
  __shared__ float tile[16][769];
  __shared__ float psum[16][16];
  __shared__ float psq[16][16];
  __shared__ float smu[16], srs[16];
  const int t  = threadIdx.x;
  const int b  = blockIdx.x / (S / 16);
  const int s0 = (blockIdx.x % (S / 16)) * 16;

  for (int r = 0; r < 16; ++r)
    for (int j = 0; j < 3; ++j) {
      int c = t + (j << 8);
      tile[r][c] = x[((size_t)(b * S + s0 + r)) * D + c];
    }
  __syncthreads();

  const int r = t >> 4, c16 = t & 15;
  float sm = 0.f, sq = 0.f;
  for (int i = 0; i < 48; ++i) {
    float v = tile[r][c16 + (i << 4)];
    sm += v; sq += v * v;
  }
  psum[r][c16] = sm; psq[r][c16] = sq;
  __syncthreads();
  if (t < 16) {
    float a = 0.f, q = 0.f;
    for (int i = 0; i < 16; ++i) { a += psum[t][i]; q += psq[t][i]; }
    float mu = a * (1.0f / 768.0f);
    float var = q * (1.0f / 768.0f) - mu * mu;
    smu[t] = mu; srs[t] = rsqrtf(var + EPS);
  }
  __syncthreads();

  for (int i = 0; i < 48; ++i) {
    int idx = t + (i << 8);
    int d = idx >> 4, sl = idx & 15;
    float v = (tile[sl][d] - smu[sl]) * srs[sl] * gamma[d] + beta[d];
    xnT[((size_t)(b * D + d)) * S + s0 + sl] = v;
  }
}

// ================= shared-memory radix-2 DIT FFT, N=8192 =================
__device__ void fft8192(float* re, float* im, float sgn) {
  const int t = threadIdx.x;
  for (int len = 2; len <= 8192; len <<= 1) {
    const int half = len >> 1;
    const float ang = sgn * 6.283185307179586f / (float)len;
    for (int i = t; i < 4096; i += 256) {
      int j = i & (half - 1);
      int pos = (i << 1) - j;
      float s, c;
      __sincosf(ang * (float)j, &s, &c);
      float ure = re[pos],        uim = im[pos];
      float vre = re[pos + half], vim = im[pos + half];
      float tre = vre * c - vim * s;
      float tim = vre * s + vim * c;
      re[pos] = ure + tre;        im[pos] = uim + tim;
      re[pos + half] = ure - tre; im[pos + half] = uim - tim;
    }
    __syncthreads();
  }
}

// ================= 2) rFFT: two real columns per complex FFT =================
__global__ __launch_bounds__(256) void rfft_kernel(const float* __restrict__ xnT,
                                                   float* __restrict__ Xre,
                                                   float* __restrict__ Xim) {
  extern __shared__ float smem[];
  float* re = smem;
  float* im = smem + 8192;
  const int t  = threadIdx.x;
  const int dp = blockIdx.x % (D / 2);
  const int b  = blockIdx.x / (D / 2);
  const int d0 = dp * 2;

  const float* p0 = xnT + ((size_t)(b * D + d0)) * S;
  const float* p1 = p0 + S;
  for (int s = t; s < 8192; s += 256) {
    int r = __brev(s) >> 19;
    re[r] = p0[s];
    im[r] = p1[s];
  }
  __syncthreads();
  fft8192(re, im, -1.0f);

  float* o0re = Xre + ((size_t)(b * D + d0)) * F;
  float* o1re = o0re + F;
  float* o0im = Xim + ((size_t)(b * D + d0)) * F;
  float* o1im = o0im + F;
  for (int k = t; k <= 4096; k += 256) {
    int nk = (8192 - k) & 8191;
    float ar = re[k], ai = im[k], br = re[nk], bi = im[nk];
    o0re[k] = 0.5f * (ar + br) * ORTHO;
    o0im[k] = 0.5f * (ai - bi) * ORTHO;
    o1re[k] = 0.5f * (ai + bi) * ORTHO;
    o1im[k] = 0.5f * (br - ar) * ORTHO;
  }
}

// ================= 3) importance-weighted magnitude scores =================
__global__ __launch_bounds__(256) void scores_kernel(const float* __restrict__ Xre,
                                                     const float* __restrict__ Xim,
                                                     const float* __restrict__ fimp,
                                                     float* __restrict__ scores) {
  __shared__ float imp[HD];
  const int row = blockIdx.y;               // b*H + h
  const int b = row / H, h = row % H;
  if (threadIdx.x < HD) imp[threadIdx.x] = sigm(fimp[h * HD + threadIdx.x]);
  __syncthreads();
  const int f = blockIdx.x * 256 + threadIdx.x;
  if (f < F) {
    float acc = 0.f;
    size_t base = ((size_t)(b * D + h * HD)) * F + f;
    for (int j = 0; j < HD; ++j) {
      float re = Xre[base + (size_t)j * F];
      float im = Xim[base + (size_t)j * F];
      acc += sqrtf(re * re + im * im) * imp[j];
    }
    scores[(size_t)row * F + f] = acc * (1.0f / (float)HD);
  }
}

// ================= 4) exact top-k mask via radix binary search =================
__global__ __launch_bounds__(256) void topk_kernel(const float* __restrict__ scores,
                                                   float* __restrict__ mask) {
  __shared__ float arr[F];
  __shared__ int redi[256];
  const int t = threadIdx.x;
  const int row = blockIdx.x;
  const float* sp = scores + (size_t)row * F;
  for (int f = t; f < F; f += 256) arr[f] = sp[f];
  __syncthreads();

  unsigned lo = 0u, hi = 0x7f800000u;
  while (lo < hi) {
    unsigned mid = (lo + hi) >> 1;
    float mv = __uint_as_float(mid);
    int c = 0;
    for (int f = t; f < F; f += 256) c += (arr[f] > mv) ? 1 : 0;
    redi[t] = c;
    __syncthreads();
    for (int off = 128; off > 0; off >>= 1) {
      if (t < off) redi[t] += redi[t + off];
      __syncthreads();
    }
    int cnt = redi[0];
    __syncthreads();
    if (cnt < KTOP) hi = mid; else lo = mid + 1;
  }
  const float T = __uint_as_float(lo);

  int c = 0;
  for (int f = t; f < F; f += 256) c += (arr[f] > T) ? 1 : 0;
  redi[t] = c;
  __syncthreads();
  for (int off = 128; off > 0; off >>= 1) {
    if (t < off) redi[t] += redi[t + off];
    __syncthreads();
  }
  const int cgt = redi[0];
  const int m = KTOP - cgt;

  float* mp = mask + (size_t)row * F;
  for (int f = t; f < F; f += 256) {
    float v = arr[f];
    float o = 0.f;
    if (v > T) {
      o = 1.f;
    } else if (v == T) {
      int rank = 0;
      for (int f2 = 0; f2 < f; ++f2) rank += (arr[f2] == T) ? 1 : 0;
      if (rank < m) o = 1.f;
    }
    mp[f] = o;
  }
}

// ================= 5) apply mask + sigmoid head gate =================
__global__ __launch_bounds__(256) void gate_kernel(float* __restrict__ Xre,
                                                   float* __restrict__ Xim,
                                                   const float* __restrict__ mask,
                                                   const float* __restrict__ hw) {
  size_t idx = (size_t)blockIdx.x * 256 + threadIdx.x;
  int bd = (int)(idx / F);
  int f  = (int)(idx - (size_t)bd * F);
  int b  = bd / D;
  int d  = bd - b * D;
  int h  = d >> 6;
  float g = sigm(hw[d]);
  float mv = mask[((size_t)(b * H + h)) * F + f];
  float sc = g * mv;
  Xre[idx] *= sc;
  Xim[idx] *= sc;
}

// ========= 6) irFFT: hermitian rebuild, inverse FFT, write y as bf16 =========
__global__ __launch_bounds__(256) void irfft_kernel(const float* __restrict__ Xre,
                                                    const float* __restrict__ Xim,
                                                    __bf16* __restrict__ ybfT) {
  extern __shared__ float smem[];
  float* re = smem;
  float* im = smem + 8192;
  const int t  = threadIdx.x;
  const int dp = blockIdx.x % (D / 2);
  const int b  = blockIdx.x / (D / 2);
  const int d0 = dp * 2;

  const float* i0re = Xre + ((size_t)(b * D + d0)) * F;
  const float* i1re = i0re + F;
  const float* i0im = Xim + ((size_t)(b * D + d0)) * F;
  const float* i1im = i0im + F;

  for (int k = t; k < 8192; k += 256) {
    float zr, zi;
    if (k <= 4096) {
      zr = i0re[k] - i1im[k];
      zi = i0im[k] + i1re[k];
    } else {
      int mm = 8192 - k;
      zr = i0re[mm] + i1im[mm];
      zi = i1re[mm] - i0im[mm];
    }
    int r = __brev(k) >> 19;
    re[r] = zr;
    im[r] = zi;
  }
  __syncthreads();
  fft8192(re, im, +1.0f);

  __bf16* y0 = ybfT + ((size_t)(b * D + d0)) * S;
  __bf16* y1 = y0 + S;
  for (int s = t; s < 8192; s += 256) {
    y0[s] = f2bf(re[s] * ORTHO);
    y1[s] = f2bf(im[s] * ORTHO);
  }
}

// ================= 6b) W (D,D) fp32 -> bf16 =================
__global__ __launch_bounds__(256) void wconv_kernel(const float* __restrict__ W,
                                                    __bf16* __restrict__ Wbf) {
  size_t i = (size_t)blockIdx.x * 256 + threadIdx.x;   // D*D = 2304*256 exactly
  Wbf[i] = f2bf(W[i]);
}

// === 6c) bf16 transpose (B,D,S)->(B,S,D) via async global->LDS DMA tiles ===
__global__ __launch_bounds__(256) void transpose_bf_kernel(const __bf16* __restrict__ ybfT,
                                                           __bf16* __restrict__ ySD) {
  __shared__ __bf16 tile[32][40];     // rows 80B (16B aligned) for b128 async writes
  const int t  = threadIdx.x;
  const int b  = blockIdx.z;
  const int d0 = blockIdx.y * 32;
  const int s0 = blockIdx.x * 32;

  // stage 32(d)x16(s) bf16 tile: 64 lane-transfers of 16B, async DMA into LDS
  if (t < 64) {
    int r  = t >> 1;                  // d-row in tile
    int ch = t & 1;                   // 16B chunk along s
    unsigned lds  = (unsigned)(uintptr_t)&tile[r][ch * 8];
    unsigned voff = (unsigned)((((size_t)(b * D + d0 + r)) * S + s0) * 2 + ch * 16);
    asm volatile("global_load_async_to_lds_b128 %0, %1, %2"
                 :: "v"(lds), "v"(voff), "s"(ybfT)
                 : "memory");
  }
  asm volatile("s_wait_asynccnt 0x0" ::: "memory");
  __syncthreads();

  // write transposed: consecutive lanes -> consecutive d (coalesced)
  for (int i = 0; i < 4; ++i) {
    int srow = (t >> 5) + i * 8;
    int dcol = t & 31;
    ySD[((size_t)(b * S + s0 + srow)) * D + d0 + dcol] = tile[dcol][srow];
  }
}

// ========== 7) LDS-free bf16-WMMA GEMM: out = residual + y @ W^T + b ==========
// A (y, (B,S,D) bf16) and B (Wbf, (D,D) bf16) are K-contiguous, so each WMMA
// fragment is exactly two global_load_b128 per lane. L2 (192MB) serves reuse.
__global__ __launch_bounds__(256) void gemm_kernel(const __bf16* __restrict__ ySD,
                                                   const __bf16* __restrict__ Wbf,
                                                   const float* __restrict__ bout,
                                                   const float* __restrict__ xres,
                                                   float* __restrict__ out) {
  const int t    = threadIdx.x;
  const int b    = blockIdx.z;
  const int ms0  = blockIdx.x * 128;
  const int n0   = blockIdx.y * 128;
  const int wave = t >> 5;
  const int lane = t & 31;
  const int wm   = (wave >> 2) * 64;
  const int wn   = (wave & 3) * 32;
  const int l15  = lane & 15;
  const int hi   = lane >> 4;

  v8f acc[4][2];
  for (int mi = 0; mi < 4; ++mi)
    for (int ni = 0; ni < 2; ++ni)
      for (int r = 0; r < 8; ++r) acc[mi][ni][r] = 0.f;

  // per-lane base pointers (K-offset parts hoisted; advance by 32 each step)
  const __bf16* ap[4];
#pragma unroll
  for (int mi = 0; mi < 4; ++mi)
    ap[mi] = ySD + ((size_t)(b * S + ms0 + wm + mi * 16 + l15)) * D + hi * 8;
  const __bf16* bp[2];
#pragma unroll
  for (int ni = 0; ni < 2; ++ni)
    bp[ni] = Wbf + (size_t)(n0 + wn + ni * 16 + l15) * D + hi * 8;

  for (int k0 = 0; k0 < D; k0 += 32) {
    v16bf afr[4], bfr[2];
#pragma unroll
    for (int mi = 0; mi < 4; ++mi) {
      U32x8 u;
      u.lo = gload16(ap[mi]);                  // K = k0+hi*8 .. +7
      u.hi = gload16(ap[mi] + 16);             // K = k0+hi*8+16 .. +23
      afr[mi] = __builtin_bit_cast(v16bf, u);
      ap[mi] += 32;
    }
#pragma unroll
    for (int ni = 0; ni < 2; ++ni) {
      U32x8 u;
      u.lo = gload16(bp[ni]);                  // K = k0+hi*8 .. +7
      u.hi = gload16(bp[ni] + 16);             // K = k0+hi*8+16 .. +23
      bfr[ni] = __builtin_bit_cast(v16bf, u);
      bp[ni] += 32;
    }
#pragma unroll
    for (int mi = 0; mi < 4; ++mi)
#pragma unroll
      for (int ni = 0; ni < 2; ++ni)
        acc[mi][ni] = __builtin_amdgcn_wmma_f32_16x16x32_bf16(
            false, afr[mi], false, bfr[ni], (short)0, acc[mi][ni], false, false);
  }

  // epilogue: C/D layout -> VGPR r holds M = r + 8*hi, N = l15
#pragma unroll
  for (int mi = 0; mi < 4; ++mi)
#pragma unroll
    for (int ni = 0; ni < 2; ++ni)
#pragma unroll
      for (int r = 0; r < 8; ++r) {
        int m = ms0 + wm + mi * 16 + r + (hi << 3);
        int n = n0 + wn + ni * 16 + l15;
        size_t o = ((size_t)(b * S + m)) * D + n;
        out[o] = xres[o] + bout[n] + acc[mi][ni][r];
      }
}

// ================= host-side orchestration =================
extern "C" void kernel_launch(void* const* d_in, const int* in_sizes, int n_in,
                              void* d_out, int out_size, void* d_ws, size_t ws_size,
                              hipStream_t stream) {
  (void)in_sizes; (void)n_in; (void)out_size; (void)ws_size;
  const float* x     = (const float*)d_in[0];
  const float* gamma = (const float*)d_in[1];
  const float* beta  = (const float*)d_in[2];
  const float* fimp  = (const float*)d_in[3];
  const float* hw    = (const float*)d_in[4];
  const float* W     = (const float*)d_in[5];
  const float* bout  = (const float*)d_in[6];
  float* out = (float*)d_out;

  float* ws     = (float*)d_ws;
  float* xnT    = ws;                    // NBDS f32; region reused for bf16 y below
  float* Xre    = ws + NBDS;             // (B,D,F)
  float* Xim    = Xre + NBDF;            // (B,D,F)
  float* scores = Xim + NBDF;            // (B,H,F)
  float* maskp  = scores + BHF;          // (B,H,F)
  __bf16* Wbf   = (__bf16*)(maskp + BHF);        // D*D bf16
  __bf16* ybfT  = (__bf16*)xnT;                  // NBDS bf16 (first half of region)
  __bf16* ySD   = ybfT + NBDS;                   // NBDS bf16 (second half of region)

  ln_kernel<<<dim3(B * (S / 16)), 256, 0, stream>>>(x, gamma, beta, xnT);
  rfft_kernel<<<dim3(B * (D / 2)), 256, 65536, stream>>>(xnT, Xre, Xim);
  scores_kernel<<<dim3((F + 255) / 256, B * H), 256, 0, stream>>>(Xre, Xim, fimp, scores);
  topk_kernel<<<dim3(B * H), 256, 0, stream>>>(scores, maskp);
  gate_kernel<<<dim3((unsigned)(NBDF / 256)), 256, 0, stream>>>(Xre, Xim, maskp, hw);
  irfft_kernel<<<dim3(B * (D / 2)), 256, 65536, stream>>>(Xre, Xim, ybfT);
  wconv_kernel<<<dim3(D * D / 256), 256, 0, stream>>>(W, Wbf);
  transpose_bf_kernel<<<dim3(S / 32, D / 32, B), 256, 0, stream>>>(ybfT, ySD);
  gemm_kernel<<<dim3(S / 128, D / 128, B), 256, 0, stream>>>(ySD, Wbf, bout, x, out);
}
